// GatedGAT_35596688949920
// MI455X (gfx1250) — matrix operand order repaired and checked by reading, not compiled, not run on hardware
//
#include <hip/hip_runtime.h>
#include <math.h>

// Problem constants (match reference)
#define IN_F   128
#define OUT_F  64
#define MAP_F  128
#define NH     4
#define KCAT   (IN_F + OUT_F)   // 192

typedef __attribute__((ext_vector_type(16))) _Float16 v16h;
typedef __attribute__((ext_vector_type(8)))  _Float16 v8h;
typedef __attribute__((ext_vector_type(8)))  float    v8f;

// ---------------------------------------------------------------------------
// fill / convert kernels
// ---------------------------------------------------------------------------
__global__ void k_fill(float* __restrict__ p, float v, long n) {
    long i = (long)blockIdx.x * blockDim.x + threadIdx.x;
    if (i < n) p[i] = v;
}

__global__ void k_cvt(const float* __restrict__ in, _Float16* __restrict__ out, long n) {
    long i = (long)blockIdx.x * blockDim.x + threadIdx.x;
    if (i < n) out[i] = (_Float16)in[i];
}

// convert [rows,cols] f32 into f16 at column offset ooff of a row-stride
// ostride f16 matrix
__global__ void k_cvt_pad(const float* __restrict__ in, _Float16* __restrict__ out,
                          long rows, int cols, int ostride, int ooff) {
    long i = (long)blockIdx.x * blockDim.x + threadIdx.x;
    if (i >= rows * cols) return;
    long r = i / cols; int c = (int)(i - r * cols);
    out[(size_t)r * ostride + ooff + c] = (_Float16)in[i];
}

// ---------------------------------------------------------------------------
// Y[nrows,C] = A[nrows,K](f16, row stride lda) @ B[C,K](f16)^T (+ bias)
// One wave computes a 16x64 output tile: A fragment loaded once per k-step,
// 4 WMMAs per k-step. Requires: nrows % 16 == 0, K % 32 == 0, C % 64 == 0.
// ---------------------------------------------------------------------------
__global__ void k_gemm16(const _Float16* __restrict__ A, int lda,
                         const _Float16* __restrict__ B,
                         const float* __restrict__ bias, float* __restrict__ Y,
                         int nrows, int K, int C) {
    const int wid  = (blockIdx.x * blockDim.x + threadIdx.x) >> 5;
    const int lane = threadIdx.x & 31;
    const int tiles_c = C >> 6;                 // 64-wide column tiles
    const int mt = wid / tiles_c;
    const int ct = wid - mt * tiles_c;
    if (mt * 16 >= nrows) return;               // wave-uniform
    const int half = lane >> 4;
    const int l16  = lane & 15;

    const _Float16* arow = A + (size_t)(mt * 16 + l16) * lda;
    v8f acc[4] = {v8f{}, v8f{}, v8f{}, v8f{}};

    for (int k0 = 0; k0 < K; k0 += 32) {
        // 16-bit A layout: lanes 0-15: elem[0..7]=K0..7, elem[8..15]=K16..23
        //                  lanes 16-31: elem[0..7]=K8..15, elem[8..15]=K24..31
        v8h alo = *(const v8h*)(arow + k0 + half * 8);
        v8h ahi = *(const v8h*)(arow + k0 + 16 + half * 8);
        v16h a = __builtin_shufflevector(alo, ahi,
                    0,1,2,3,4,5,6,7,8,9,10,11,12,13,14,15);
        #pragma unroll
        for (int j = 0; j < 4; ++j) {
            // B layout (K x N): same K order as A (lanes 0-15: K0..7,K16..23; lanes 16-31: K8..15,K24..31)
            const _Float16* brow = B + (size_t)(ct * 64 + j * 16 + l16) * K;
            v8h blo = *(const v8h*)(brow + k0 + half * 8);
            v8h bhi = *(const v8h*)(brow + k0 + 16 + half * 8);
            v16h b = __builtin_shufflevector(blo, bhi,
                        0,1,2,3,4,5,6,7,8,9,10,11,12,13,14,15);
            acc[j] = __builtin_amdgcn_wmma_f32_16x16x32_f16(
                        false, a, false, b, (short)0, acc[j], false, false);
        }
    }

    #pragma unroll
    for (int j = 0; j < 4; ++j) {
        const int col = ct * 64 + j * 16 + l16;
        const float bv = bias ? bias[col] : 0.0f;
        float* yout = Y + (size_t)(mt * 16 + half * 8) * C + col;
        #pragma unroll
        for (int v = 0; v < 8; ++v)
            yout[(size_t)v * C] = acc[j][v] + bv;
    }
}

// ---------------------------------------------------------------------------
// el/er: per (node,head) dot of feat with attn_l / attn_r
// ---------------------------------------------------------------------------
__global__ void k_el_er(const float* __restrict__ feat,
                        const float* __restrict__ attn_l, const float* __restrict__ attn_r,
                        float* __restrict__ el, float* __restrict__ er, int N) {
    int t = blockIdx.x * blockDim.x + threadIdx.x;
    if (t >= N * NH) return;
    int n = t >> 2, h = t & 3;
    const float* f  = feat + (size_t)n * (NH * OUT_F) + h * OUT_F;
    const float* al = attn_l + h * OUT_F;
    const float* ar = attn_r + h * OUT_F;
    float sl = 0.f, sr = 0.f;
    #pragma unroll 8
    for (int d = 0; d < OUT_F; ++d) { float fv = f[d]; sl += fv * al[d]; sr += fv * ar[d]; }
    el[t] = sl; er[t] = sr;
}

// ---------------------------------------------------------------------------
// gate-path mailboxes: deg, sum_z (mean numerator), max_z
// thread = (edge, 4-feature chunk); 32 chunks cover 128 features
// ---------------------------------------------------------------------------
__global__ void k_edge_gather(const int* __restrict__ src, const int* __restrict__ dstv,
                              const float* __restrict__ x, const float* __restrict__ mz,
                              float* __restrict__ deg, float* __restrict__ sum_z,
                              float* __restrict__ max_z, int E) {
    long tid = (long)blockIdx.x * blockDim.x + threadIdx.x;
    long e = tid >> 5;
    if (e >= E) return;
    int chunk = (int)(tid & 31);
    int s = src[e], d = dstv[e];
    int f = chunk * 4;
    float4 xs = *(const float4*)&x [(size_t)s * IN_F  + f];
    float4 ms = *(const float4*)&mz[(size_t)s * MAP_F + f];
    float* sz  = &sum_z[(size_t)d * IN_F  + f];
    float* mxz = &max_z[(size_t)d * MAP_F + f];
    atomicAdd(sz + 0, xs.x); atomicAdd(sz + 1, xs.y);
    atomicAdd(sz + 2, xs.z); atomicAdd(sz + 3, xs.w);
    __hip_atomic_fetch_max(mxz + 0, ms.x, __ATOMIC_RELAXED, __HIP_MEMORY_SCOPE_AGENT);
    __hip_atomic_fetch_max(mxz + 1, ms.y, __ATOMIC_RELAXED, __HIP_MEMORY_SCOPE_AGENT);
    __hip_atomic_fetch_max(mxz + 2, ms.z, __ATOMIC_RELAXED, __HIP_MEMORY_SCOPE_AGENT);
    __hip_atomic_fetch_max(mxz + 3, ms.w, __ATOMIC_RELAXED, __HIP_MEMORY_SCOPE_AGENT);
    if (chunk == 0) atomicAdd(&deg[d], 1.0f);
}

__device__ __forceinline__ float leaky(float t) { return t > 0.f ? t : 0.2f * t; }

// edge softmax pass 1: per-(dst,head) max of leaky_relu(el[src]+er[dst])
__global__ void k_edge_emax(const int* __restrict__ src, const int* __restrict__ dstv,
                            const float* __restrict__ el, const float* __restrict__ er,
                            float* __restrict__ emax, int E) {
    long e = (long)blockIdx.x * blockDim.x + threadIdx.x;
    if (e >= E) return;
    int s = src[e], d = dstv[e];
    #pragma unroll
    for (int h = 0; h < NH; ++h) {
        float t = leaky(el[s * NH + h] + er[d * NH + h]);
        __hip_atomic_fetch_max(&emax[d * NH + h], t, __ATOMIC_RELAXED, __HIP_MEMORY_SCOPE_AGENT);
    }
}

// edge softmax pass 2: per-(dst,head) sum of exp(e - emax)
__global__ void k_edge_esum(const int* __restrict__ src, const int* __restrict__ dstv,
                            const float* __restrict__ el, const float* __restrict__ er,
                            const float* __restrict__ emax, float* __restrict__ esum, int E) {
    long e = (long)blockIdx.x * blockDim.x + threadIdx.x;
    if (e >= E) return;
    int s = src[e], d = dstv[e];
    #pragma unroll
    for (int h = 0; h < NH; ++h) {
        float t  = leaky(el[s * NH + h] + er[d * NH + h]);
        float mx = emax[d * NH + h]; if (!(mx > -1e38f)) mx = 0.f;
        atomicAdd(&esum[d * NH + h], __expf(t - mx));
    }
}

// weighted aggregation: rst[dst,h,:] += a * feat[src,h,:]
// thread = (edge, head)
__global__ void k_edge_rst(const int* __restrict__ src, const int* __restrict__ dstv,
                           const float* __restrict__ el, const float* __restrict__ er,
                           const float* __restrict__ emax, const float* __restrict__ esum,
                           const float* __restrict__ weight, const float* __restrict__ feat,
                           float* __restrict__ rst, int E) {
    long tid = (long)blockIdx.x * blockDim.x + threadIdx.x;
    long e = tid >> 2;
    if (e >= E) return;
    int h = (int)(tid & 3);
    int s = src[e], d = dstv[e];
    float t  = leaky(el[s * NH + h] + er[d * NH + h]);
    float mx = emax[d * NH + h]; if (!(mx > -1e38f)) mx = 0.f;
    float a = __expf(t - mx) / esum[d * NH + h] * weight[e];
    const float* f = feat + (size_t)s * (NH * OUT_F) + h * OUT_F;
    float*       r = rst  + (size_t)d * (NH * OUT_F) + h * OUT_F;
    #pragma unroll
    for (int i = 0; i < OUT_F; i += 4) {
        float4 fv = *(const float4*)&f[i];
        atomicAdd(&r[i + 0], a * fv.x);
        atomicAdd(&r[i + 1], a * fv.y);
        atomicAdd(&r[i + 2], a * fv.z);
        atomicAdd(&r[i + 3], a * fv.w);
    }
}

// gate = sigmoid(concat([x, max_z, mean_z]) @ gate_fn_w.T + gate_fn_b)
__global__ void k_gate(const float* __restrict__ x, const float* __restrict__ max_z,
                       const float* __restrict__ sum_z, const float* __restrict__ deg,
                       const float* __restrict__ gw, const float* __restrict__ gb,
                       float* __restrict__ gate, int N) {
    int t = blockIdx.x * blockDim.x + threadIdx.x;
    if (t >= N * NH) return;
    int n = t >> 2, h = t & 3;
    const float* w   = gw + (size_t)h * (2 * IN_F + MAP_F);
    const float* xr  = x     + (size_t)n * IN_F;
    const float* mzr = max_z + (size_t)n * MAP_F;
    const float* szr = sum_z + (size_t)n * IN_F;
    float invd = 1.0f / fmaxf(deg[n], 1.0f);
    float acc = gb[h];
    #pragma unroll 8
    for (int k = 0; k < IN_F; ++k)  acc += xr[k] * w[k];
    #pragma unroll 8
    for (int k = 0; k < MAP_F; ++k) { float m = mzr[k]; if (!(m > -1e38f)) m = 0.f;
                                      acc += m * w[IN_F + k]; }
    #pragma unroll 8
    for (int k = 0; k < IN_F; ++k)  acc += szr[k] * invd * w[IN_F + MAP_F + k];
    gate[t] = 1.0f / (1.0f + __expf(-acc));
}

// gated = mean over heads of gate * rst
__global__ void k_gated(const float* __restrict__ gate, const float* __restrict__ rst,
                        float* __restrict__ gated, int N) {
    int t = blockIdx.x * blockDim.x + threadIdx.x;
    if (t >= N * OUT_F) return;
    int n = t / OUT_F, dd = t - n * OUT_F;
    float acc = 0.f;
    #pragma unroll
    for (int h = 0; h < NH; ++h)
        acc += gate[n * NH + h] * rst[(size_t)n * (NH * OUT_F) + h * OUT_F + dd];
    gated[t] = acc * 0.25f;
}

// ---------------------------------------------------------------------------
// launch
// ---------------------------------------------------------------------------
extern "C" void kernel_launch(void* const* d_in, const int* in_sizes, int n_in,
                              void* d_out, int out_size, void* d_ws, size_t ws_size,
                              hipStream_t stream) {
    const float* x         = (const float*)d_in[0];
    const int*   src       = (const int*)  d_in[1];
    const int*   dstv      = (const int*)  d_in[2];
    const float* weight    = (const float*)d_in[3];
    const float* fc_w      = (const float*)d_in[4];
    const float* attn_l    = (const float*)d_in[5];
    const float* attn_r    = (const float*)d_in[6];
    const float* gate_m_w  = (const float*)d_in[7];
    const float* gate_m_b  = (const float*)d_in[8];
    const float* gate_fn_w = (const float*)d_in[9];
    const float* gate_fn_b = (const float*)d_in[10];
    const float* merger_w  = (const float*)d_in[11];
    const float* merger_b  = (const float*)d_in[12];
    float* out = (float*)d_out;

    const int N = in_sizes[0] / IN_F;   // 50000 (multiple of 16)
    const int E = in_sizes[1];          // 800000

    // -------- workspace layout (f32 region, then f16 region) --------
    float* p = (float*)d_ws;
    float* mz    = p; p += (size_t)N * MAP_F;
    float* feat  = p; p += (size_t)N * NH * OUT_F;
    float* el    = p; p += (size_t)N * NH;
    float* er    = p; p += (size_t)N * NH;
    float* deg   = p; p += (size_t)N;            // deg, sum_z contiguous (zero)
    float* sum_z = p; p += (size_t)N * IN_F;
    float* max_z = p; p += (size_t)N * MAP_F;    // max_z, emax contiguous (-inf)
    float* emax  = p; p += (size_t)N * NH;
    float* esum  = p; p += (size_t)N * NH;       // esum, rst contiguous (zero)
    float* rst   = p; p += (size_t)N * NH * OUT_F;
    float* gate  = p; p += (size_t)N * NH;
    float* gated = p; p += (size_t)N * OUT_F;
    _Float16* h = (_Float16*)p;                  // 16B-aligned (all counts %8==0)
    _Float16* xg16   = h; h += (size_t)N * KCAT;        // [N,192] = concat(x, gated)
    _Float16* wmz16  = h; h += (size_t)MAP_F * IN_F;    // gate_m_w
    _Float16* wfc16  = h; h += (size_t)NH * OUT_F * IN_F; // fc_w
    _Float16* wmg16  = h; h += (size_t)OUT_F * KCAT;    // merger_w

    const float NEG_INF = -__builtin_inff();
    auto blocks1d = [](long n, int bs) { return (int)((n + bs - 1) / bs); };

    // -------- init accumulators (every call: deterministic for graph replay) --
    {
        long nz1 = (long)N * (1 + IN_F);                 // deg + sum_z
        long ni  = (long)N * (MAP_F + NH);               // max_z + emax
        long nz2 = (long)N * (NH + NH * OUT_F);          // esum + rst
        k_fill<<<blocks1d(nz1, 256), 256, 0, stream>>>(deg,   0.0f,    nz1);
        k_fill<<<blocks1d(ni,  256), 256, 0, stream>>>(max_z, NEG_INF, ni);
        k_fill<<<blocks1d(nz2, 256), 256, 0, stream>>>(esum,  0.0f,    nz2);
    }

    // -------- one-shot f32 -> f16 conversions --------
    k_cvt<<<blocks1d((long)MAP_F * IN_F, 256), 256, 0, stream>>>(gate_m_w, wmz16, (long)MAP_F * IN_F);
    k_cvt<<<blocks1d((long)NH * OUT_F * IN_F, 256), 256, 0, stream>>>(fc_w, wfc16, (long)NH * OUT_F * IN_F);
    k_cvt<<<blocks1d((long)OUT_F * KCAT, 256), 256, 0, stream>>>(merger_w, wmg16, (long)OUT_F * KCAT);
    k_cvt_pad<<<blocks1d((long)N * IN_F, 256), 256, 0, stream>>>(x, xg16, N, IN_F, KCAT, 0);

    const int mtiles = (N + 15) / 16;

    // mz = x @ gate_m_w^T + gate_m_b        [N,128]
    {
        long thr = (long)mtiles * (MAP_F / 64) * 32;
        k_gemm16<<<blocks1d(thr, 256), 256, 0, stream>>>(
            xg16, KCAT, wmz16, gate_m_b, mz, N, IN_F, MAP_F);
    }
    // feat = x @ fc_w^T                     [N,256]
    {
        long thr = (long)mtiles * ((NH * OUT_F) / 64) * 32;
        k_gemm16<<<blocks1d(thr, 256), 256, 0, stream>>>(
            xg16, KCAT, wfc16, nullptr, feat, N, IN_F, NH * OUT_F);
    }
    // el/er
    k_el_er<<<blocks1d((long)N * NH, 256), 256, 0, stream>>>(feat, attn_l, attn_r, el, er, N);

    // gate mailboxes
    k_edge_gather<<<blocks1d((long)E * 32, 256), 256, 0, stream>>>(
        src, dstv, x, mz, deg, sum_z, max_z, E);

    // edge softmax
    k_edge_emax<<<blocks1d(E, 256), 256, 0, stream>>>(src, dstv, el, er, emax, E);
    k_edge_esum<<<blocks1d(E, 256), 256, 0, stream>>>(src, dstv, el, er, emax, esum, E);
    k_edge_rst <<<blocks1d((long)E * NH, 256), 256, 0, stream>>>(
        src, dstv, el, er, emax, esum, weight, feat, rst, E);

    // gate + gated merge
    k_gate <<<blocks1d((long)N * NH, 256), 256, 0, stream>>>(
        x, max_z, sum_z, deg, gate_fn_w, gate_fn_b, gate, N);
    k_gated<<<blocks1d((long)N * OUT_F, 256), 256, 0, stream>>>(gate, rst, gated, N);

    // concat: gated -> xg16 columns 128..191, then
    // out = concat([x, gated]) @ merger_w^T + merger_b
    k_cvt_pad<<<blocks1d((long)N * OUT_F, 256), 256, 0, stream>>>(gated, xg16, N, OUT_F, KCAT, IN_F);
    {
        long thr = (long)mtiles * (OUT_F / 64) * 32;
        k_gemm16<<<blocks1d(thr, 256), 256, 0, stream>>>(
            xg16, KCAT, wmg16, merger_b, out, N, KCAT, OUT_F);
    }
}
